// MIFM_75453985457477
// MI455X (gfx1250) — compile-verified, matrix operand order checked by reading, not run on hardware
//
#include <hip/hip_runtime.h>
#include <hip/hip_bf16.h>

// ---------------------------------------------------------------------------
// Problem constants (from reference): B=8, C=256, H=W=128, P=7, N=500, HID=16
// ---------------------------------------------------------------------------
constexpr int B_  = 8;
constexpr int C_  = 256;
constexpr int H_  = 128;
constexpr int W_  = 128;
constexpr int HW  = H_ * W_;        // 16384
constexpr int N_  = 500;
constexpr int NP  = 512;            // padded N (multiple of 16 for WMMA tiles)
constexpr int P_  = 7;
constexpr int PP  = P_ * P_;        // 49
constexpr int HID = 16;

typedef __attribute__((ext_vector_type(16))) __bf16 bf16x16;
typedef __attribute__((ext_vector_type(8)))  __bf16 bf16x8;
typedef __attribute__((ext_vector_type(8)))  float  f32x8;

// CDNA5 16-bit WMMA A/B operand layout (16x32 tile), lane half h:
//   halfword j in [0,16):  K = (j>=8 ? 16 : 0) + h*8 + (j&7)
// => per kk-step of 32 K-values, the 16 halfwords are two contiguous 8-element
//    runs starting at K = kk*32 + h*8 and K = kk*32 + 16 + h*8.
// With 8-bf16 groups (16 B) over a 512/256-element row: group = kk*4 + h and
// kk*4 + 2 + h.
__device__ __forceinline__ bf16x16 gather_op(const bf16x8* __restrict__ row,
                                             int kk, int half) {
  bf16x8 lo = row[kk * 4 + half];
  bf16x8 hi = row[kk * 4 + 2 + half];
  return __builtin_shufflevector(lo, hi, 0, 1, 2, 3, 4, 5, 6, 7,
                                 8, 9, 10, 11, 12, 13, 14, 15);
}

__device__ __forceinline__ float sigmoidf_(float x) {
  return 1.0f / (1.0f + __expf(-x));
}

// ---------------------------------------------------------------------------
// 1. Saliency score: sigmoid(<x_main[b,:,h,w], scorer_w> + scorer_b)
// ---------------------------------------------------------------------------
__global__ __launch_bounds__(256) void score_kernel(
    const float* __restrict__ x, const float* __restrict__ sw,
    const float* __restrict__ sb, float* __restrict__ score) {
  int gid = blockIdx.x * 256 + threadIdx.x;   // over B*HW
  int b   = gid >> 14;                        // /16384
  int pix = gid & (HW - 1);
  const float* xp = x + (size_t)b * C_ * HW + pix;
  float s = sb[0];
#pragma unroll 8
  for (int c = 0; c < C_; ++c) s += xp[(size_t)c * HW] * sw[c];
  score[gid] = sigmoidf_(s);
}

// ---------------------------------------------------------------------------
// 2. Top-k (k=500) per batch, iterative block-wide argmax.
//    Key = (monotone(float) << 32) | (16383 - pos)  => max value, min index.
// ---------------------------------------------------------------------------
__global__ __launch_bounds__(1024) void topk_kernel(
    const float* __restrict__ score, int* __restrict__ idx) {
  __shared__ unsigned long long red[1024];
  __shared__ unsigned int taken[HW / 32];     // 2 KB bitmask
  const int b = blockIdx.x;
  const int t = threadIdx.x;
  const float* s = score + (size_t)b * HW;
  for (int w = t; w < HW / 32; w += 1024) taken[w] = 0u;
  __syncthreads();
  for (int i = 0; i < N_; ++i) {
    unsigned long long best = 0ull;
    for (int p = t; p < HW; p += 1024) {
      if ((taken[p >> 5] >> (p & 31)) & 1u) continue;
      unsigned int vb = __float_as_uint(s[p]);
      vb = (vb & 0x80000000u) ? ~vb : (vb | 0x80000000u);  // monotone map
      unsigned long long key =
          ((unsigned long long)vb << 32) | (unsigned int)(HW - 1 - p);
      if (key > best) best = key;
    }
    red[t] = best;
    __syncthreads();
    for (int off = 512; off > 0; off >>= 1) {
      if (t < off && red[t + off] > red[t]) red[t] = red[t + off];
      __syncthreads();
    }
    if (t == 0) {
      int pos = HW - 1 - (int)(red[0] & 0xFFFFFFFFu);
      idx[b * NP + i] = pos;
      taken[pos >> 5] |= 1u << (pos & 31);
    }
    __syncthreads();
  }
}

// ---------------------------------------------------------------------------
// 3. Patch sampling: bilinear (zero-pad) 7x7 patch, pool 0.5*(mean+max).
//    One block per (n,b), n over padded NP; thread = channel.
//    Emits: v_main f32 [B,NP,C], vm_bf [B,NP,C], va_bf [B,NP,C],
//           va_bfT [B,C,NP] (zero rows for n >= 500).
// ---------------------------------------------------------------------------
__device__ __forceinline__ float fetch_(const float* f, int y, int x) {
  if ((unsigned)y >= (unsigned)H_ || (unsigned)x >= (unsigned)W_) return 0.0f;
  return f[y * W_ + x];
}
__device__ __forceinline__ float bilin_(const float* f, int x0, int y0,
                                        float wx, float wy) {
  float g00 = fetch_(f, y0, x0),     g01 = fetch_(f, y0, x0 + 1);
  float g10 = fetch_(f, y0 + 1, x0), g11 = fetch_(f, y0 + 1, x0 + 1);
  return g00 * (1.f - wy) * (1.f - wx) + g01 * (1.f - wy) * wx +
         g10 * wy * (1.f - wx)         + g11 * wy * wx;
}

__global__ __launch_bounds__(256) void sample_kernel(
    const float* __restrict__ xm, const float* __restrict__ xa,
    const int* __restrict__ idx, float* __restrict__ vmain,
    __bf16* __restrict__ vm_bf, __bf16* __restrict__ va_bf,
    __bf16* __restrict__ va_bfT) {
  __shared__ int   sx0[PP], sy0[PP];
  __shared__ float swx[PP], swy[PP];
  const int n = blockIdx.x, b = blockIdx.y, t = threadIdx.x;
  const int c = t;
  const size_t rowo = ((size_t)b * NP + n) * C_ + c;
  const size_t colo = ((size_t)b * C_ + c) * NP + n;
  if (n >= N_) {                       // uniform across block: zero padding
    vmain[rowo]  = 0.0f;
    vm_bf[rowo]  = (__bf16)0.0f;
    va_bf[rowo]  = (__bf16)0.0f;
    va_bfT[colo] = (__bf16)0.0f;
    return;
  }
  const int pos = idx[b * NP + n];
  const int iy = pos >> 7, ix = pos & (W_ - 1);
  if (t < PP) {
    int pi = t / P_, pj = t % P_;
    float dy = (float)(pi - P_ / 2), dx = (float)(pj - P_ / 2);
    float cx = ((float)ix + 0.5f) * (2.0f / W_) - 1.0f;
    float cy = ((float)iy + 0.5f) * (2.0f / H_) - 1.0f;
    float gx = fminf(fmaxf(cx + dx * (2.0f / W_), -1.0f), 1.0f);
    float gy = fminf(fmaxf(cy + dy * (2.0f / H_), -1.0f), 1.0f);
    float px = ((gx + 1.0f) * W_ - 1.0f) * 0.5f;
    float py = ((gy + 1.0f) * H_ - 1.0f) * 0.5f;
    float x0 = floorf(px), y0 = floorf(py);
    sx0[t] = (int)x0; sy0[t] = (int)y0;
    swx[t] = px - x0; swy[t] = py - y0;
  }
  __syncthreads();
  const float* fm = xm + ((size_t)b * C_ + c) * HW;
  const float* fa = xa + ((size_t)b * C_ + c) * HW;
  float sumM = 0.f, maxM = -3.402823466e38f;
  float sumA = 0.f, maxA = -3.402823466e38f;
  for (int p = 0; p < PP; ++p) {
    int x0 = sx0[p], y0 = sy0[p];
    float wx = swx[p], wy = swy[p];
    float vM = bilin_(fm, x0, y0, wx, wy);
    float vA = bilin_(fa, x0, y0, wx, wy);
    sumM += vM; maxM = fmaxf(maxM, vM);
    sumA += vA; maxA = fmaxf(maxA, vA);
  }
  float vM = 0.5f * (sumM * (1.0f / PP) + maxM);
  float vA = 0.5f * (sumA * (1.0f / PP) + maxA);
  vmain[rowo]  = vM;
  vm_bf[rowo]  = (__bf16)vM;
  va_bf[rowo]  = (__bf16)vA;
  va_bfT[colo] = (__bf16)vA;
}

// ---------------------------------------------------------------------------
// 4. Cross attention via WMMA (bf16 inputs, f32 accumulate, f32 softmax).
//    One block (4 wave32) per 16-row n-tile per batch. All operand loads are
//    unconditional 128-bit loads from zero-padded bf16 buffers.
// ---------------------------------------------------------------------------
__global__ __launch_bounds__(128) void attn_kernel(
    const __bf16* __restrict__ vm_bf, const __bf16* __restrict__ va_bf,
    const __bf16* __restrict__ va_bfT, float* __restrict__ vhat) {
  __shared__ float  sim[16][NP + 4];          // 33 KB (pre-softmax scores)
  __shared__ __bf16 attn_bf[16][NP];          // 16 KB (normalized weights)
  __shared__ float  red[16][8];
  __shared__ float  rowmax[16], rowsum[16];
  const int tid  = threadIdx.x;
  const int lane = tid & 31, wave = tid >> 5;
  const int half = lane >> 4, l15 = lane & 15;
  const int b = blockIdx.y;
  const int n_base = blockIdx.x * 16;

  // --- preload A tile: v_main rows [n_base, n_base+16), K = 256 ---
  const bf16x8* arow =
      (const bf16x8*)(vm_bf + ((size_t)b * NP + n_base + l15) * C_);
  bf16x16 A[8];
#pragma unroll
  for (int kk = 0; kk < 8; ++kk) A[kk] = gather_op(arow, kk, half);

  // --- sim = (v_main @ v_aux^T) / sqrt(C); 16x16 tiles over m ---
  for (int mt = wave; mt < NP / 16; mt += 4) {
    const int mcol = mt * 16 + l15;
    const bf16x8* brow = (const bf16x8*)(va_bf + ((size_t)b * NP + mcol) * C_);
    f32x8 acc = {0.f, 0.f, 0.f, 0.f, 0.f, 0.f, 0.f, 0.f};
#pragma unroll
    for (int kk = 0; kk < 8; ++kk) {
      acc = __builtin_amdgcn_wmma_f32_16x16x32_bf16(
          false, A[kk], false, gather_op(brow, kk, half), (short)0, acc,
          false, false);
    }
#pragma unroll
    for (int r = 0; r < 8; ++r)
      sim[half * 8 + r][mt * 16 + l15] = acc[r] * 0.0625f;  // 1/sqrt(256)
  }
  __syncthreads();

  // --- softmax over m (500 valid columns); 8 threads per row ---
  {
    const int row = tid >> 3, seg = tid & 7;
    const int c0 = seg * 64;
    float lm = -3.402823466e38f;
    for (int c = c0; c < c0 + 64; ++c)
      if (c < N_) lm = fmaxf(lm, sim[row][c]);
    red[row][seg] = lm;
    __syncthreads();
    if (seg == 0) {
      float mx = red[row][0];
      for (int j = 1; j < 8; ++j) mx = fmaxf(mx, red[row][j]);
      rowmax[row] = mx;
    }
    __syncthreads();
    const float mx = rowmax[row];
    float ls = 0.f;
    for (int c = c0; c < c0 + 64; ++c) {
      float e = (c < N_) ? __expf(sim[row][c] - mx) : 0.0f;
      sim[row][c] = e;
      ls += e;
    }
    red[row][seg] = ls;
    __syncthreads();
    if (seg == 0) {
      float s = 0.f;
      for (int j = 0; j < 8; ++j) s += red[row][j];
      rowsum[row] = s;
    }
    __syncthreads();
    const float inv = 1.0f / rowsum[row];
    for (int c = c0; c < c0 + 64; ++c)
      attn_bf[row][c] = (__bf16)(sim[row][c] * inv);
  }
  __syncthreads();

  // --- v_hat = attn @ v_aux; c-tiles split across waves ---
  const bf16x8* attn_row = (const bf16x8*)&attn_bf[l15][0];
  for (int ct = wave * 4; ct < wave * 4 + 4; ++ct) {
    const int c = ct * 16 + l15;
    const bf16x8* bcol = (const bf16x8*)(va_bfT + ((size_t)b * C_ + c) * NP);
    f32x8 acc = {0.f, 0.f, 0.f, 0.f, 0.f, 0.f, 0.f, 0.f};
#pragma unroll
    for (int kk = 0; kk < NP / 32; ++kk) {      // K = 512 (m dimension)
      acc = __builtin_amdgcn_wmma_f32_16x16x32_bf16(
          false, gather_op(attn_row, kk, half), false,
          gather_op(bcol, kk, half), (short)0, acc, false, false);
    }
#pragma unroll
    for (int r = 0; r < 8; ++r) {
      int nrow = n_base + half * 8 + r;
      if (nrow < N_)
        vhat[((size_t)b * NP + nrow) * C_ + c] = acc[r];
    }
  }
}

// ---------------------------------------------------------------------------
// 5. Gated fusion MLP + projection. One block per (n,b); thread = channel.
// ---------------------------------------------------------------------------
__global__ __launch_bounds__(256) void fuse_kernel(
    const float* __restrict__ vmain, const float* __restrict__ vhat,
    const float* __restrict__ fc1w, const float* __restrict__ fc1b,
    const float* __restrict__ fc2w, const float* __restrict__ fc2b,
    const float* __restrict__ projw, const float* __restrict__ projb,
    float* __restrict__ vinj) {
  __shared__ float zm[C_], zh[C_], zd[C_], hid[HID], vf[C_];
  const int n = blockIdx.x, b = blockIdx.y, t = threadIdx.x;
  const size_t base = ((size_t)b * NP + n) * C_;
  float m = vmain[base + t], h = vhat[base + t];
  zm[t] = m; zh[t] = h; zd[t] = m - h;
  __syncthreads();
  if (t < HID) {
    float s = fc1b[t];
    for (int k = 0; k < C_; ++k) {
      s += zm[k] * fc1w[(0 * C_ + k) * HID + t];
      s += zh[k] * fc1w[(1 * C_ + k) * HID + t];
      s += zd[k] * fc1w[(2 * C_ + k) * HID + t];
    }
    hid[t] = fmaxf(s, 0.0f);
  }
  __syncthreads();
  float g = fc2b[t];
#pragma unroll
  for (int j = 0; j < HID; ++j) g += hid[j] * fc2w[j * C_ + t];
  g = sigmoidf_(g);
  vf[t] = m + g * (m - h);
  __syncthreads();
  float o = projb[t];
#pragma unroll 8
  for (int k = 0; k < C_; ++k) o += vf[k] * projw[k * C_ + t];
  vinj[base + t] = o;
}

// ---------------------------------------------------------------------------
// 6. out = x_main (vectorized copy), then scatter-add v_inj at topk positions
// ---------------------------------------------------------------------------
__global__ void copy_kernel(const float4* __restrict__ src,
                            float4* __restrict__ dst, int n4) {
  int i = blockIdx.x * blockDim.x + threadIdx.x;
  int stride = gridDim.x * blockDim.x;
  for (; i < n4; i += stride) dst[i] = src[i];
}

__global__ __launch_bounds__(256) void scatter_kernel(
    const float* __restrict__ vinj, const int* __restrict__ idx,
    float* __restrict__ out) {
  const int n = blockIdx.x, b = blockIdx.y, c = threadIdx.x;
  const int pos = idx[b * NP + n];               // unique per batch -> no race
  const size_t o = ((size_t)b * C_ + c) * HW + pos;
  out[o] += vinj[((size_t)b * NP + n) * C_ + c];
}

// ---------------------------------------------------------------------------
// Host launcher
// ---------------------------------------------------------------------------
extern "C" void kernel_launch(void* const* d_in, const int* in_sizes, int n_in,
                              void* d_out, int out_size, void* d_ws,
                              size_t ws_size, hipStream_t stream) {
  (void)in_sizes; (void)n_in; (void)out_size; (void)ws_size;
  const float* x_main   = (const float*)d_in[0];
  const float* x_aux    = (const float*)d_in[1];
  const float* scorer_w = (const float*)d_in[2];
  const float* scorer_b = (const float*)d_in[3];
  const float* fc1_w    = (const float*)d_in[4];
  const float* fc1_b    = (const float*)d_in[5];
  const float* fc2_w    = (const float*)d_in[6];
  const float* fc2_b    = (const float*)d_in[7];
  const float* proj_w   = (const float*)d_in[8];
  const float* proj_b   = (const float*)d_in[9];
  // d_in[10] = n (==500, fixed by reference constants; grids must be static)
  float* out = (float*)d_out;

  // workspace layout
  constexpr size_t VROW = (size_t)B_ * NP * C_;       // 1,048,576 elements
  float* ws_f   = (float*)d_ws;
  float* score  = ws_f;                               // B*HW floats
  int*   idx    = (int*)(ws_f + (size_t)B_ * HW);     // B*NP ints
  float* v_main = ws_f + (size_t)B_ * HW + B_ * NP;   // VROW floats
  float* v_hat  = v_main + VROW;                      // VROW floats
  float* v_inj  = v_hat + VROW;                       // VROW floats
  __bf16* vm_bf  = (__bf16*)(v_inj + VROW);           // VROW bf16
  __bf16* va_bf  = vm_bf + VROW;                      // VROW bf16
  __bf16* va_bfT = va_bf + VROW;                      // VROW bf16 (transposed)

  // 1) saliency scores
  score_kernel<<<(B_ * HW) / 256, 256, 0, stream>>>(x_main, scorer_w, scorer_b,
                                                    score);
  // 2) top-k per batch
  topk_kernel<<<B_, 1024, 0, stream>>>(score, idx);
  // 3) patch sampling + pooling (+ zero padding of rows [500,512))
  sample_kernel<<<dim3(NP, B_), 256, 0, stream>>>(x_main, x_aux, idx, v_main,
                                                  vm_bf, va_bf, va_bfT);
  // 4) cross attention (WMMA)
  attn_kernel<<<dim3(NP / 16, B_), 128, 0, stream>>>(vm_bf, va_bf, va_bfT,
                                                     v_hat);
  // 5) gated fusion + projection
  fuse_kernel<<<dim3(N_, B_), 256, 0, stream>>>(v_main, v_hat, fc1_w, fc1_b,
                                                fc2_w, fc2_b, proj_w, proj_b,
                                                v_inj);
  // 6) out = x_main ; out[topk] += v_inj
  const int n4 = (B_ * C_ * HW) / 4;
  copy_kernel<<<8192, 256, 0, stream>>>((const float4*)x_main, (float4*)out,
                                        n4);
  scatter_kernel<<<dim3(N_, B_), 256, 0, stream>>>(v_inj, idx, out);
}
